// Block_82875688944192
// MI455X (gfx1250) — matrix operand order repaired and checked by reading, not compiled, not run on hardware
//
#include <hip/hip_runtime.h>

// ---------------------------------------------------------------------------
// Transformer block for gfx1250 (MI455X): bf16 WMMA GEMMs + flash attention.
// Double-buffered async global->LDS staging (GLOBAL_LOAD_ASYNC_TO_LDS_B128,
// split-wait on ASYNCcnt) overlaps HBM fetch of tile k+1 with WMMA on tile k.
// ---------------------------------------------------------------------------

typedef __attribute__((ext_vector_type(16))) __bf16 v16bf;
typedef __attribute__((ext_vector_type(8)))  float  v8f;

#define TB 2
#define TT 2048
#define TC 1024
#define TH 16
#define THD 64
#define TM (TB * TT)   // 4096 rows

// ---------------- workspace layout (bytes) ----------------
#define OFF_WB_ATTN   (size_t)0            // [3072][1024] bf16 (transposed)
#define OFF_WB_APROJ  (size_t)6291456      // [1024][1024] bf16 (transposed)
#define OFF_WB_FC     (size_t)8388608      // [4096][1024] bf16 (transposed)
#define OFF_WB_MPROJ  (size_t)16777216     // [1024][4096] bf16 (transposed)
#define OFF_H1        (size_t)25165824     // 4096*1024 bf16
#define OFF_Q         (size_t)33554432     // [B,H,T,hd] bf16
#define OFF_K         (size_t)41943040
#define OFF_V         (size_t)50331648
#define OFF_Y         (size_t)58720256     // [B,T,C] bf16
#define OFF_XRES      (size_t)67108864     // [B,T,C] f32
#define OFF_H2        (size_t)83886080     // bf16
#define OFF_H3        (size_t)92274688     // [4096,4096] bf16

__device__ __forceinline__ float gelu_f(float x) {
    const float c = 0.7978845608028654f;
    return 0.5f * x * (1.0f + tanhf(c * (x + 0.044715f * x * x * x)));
}

// ---- gfx1250 async global->LDS copy helpers (tracked by ASYNCcnt) ----
__device__ __forceinline__ void async_copy_b128(void* lds_dst, const void* gsrc) {
    unsigned l = (unsigned)(uintptr_t)lds_dst;   // low 32 bits = LDS offset
    asm volatile("global_load_async_to_lds_b128 %0, %1, off"
                 :: "v"(l), "v"(gsrc) : "memory");
}
__device__ __forceinline__ void wait_async0() {
#if __has_builtin(__builtin_amdgcn_s_wait_asynccnt)
    __builtin_amdgcn_s_wait_asynccnt(0);
#else
    asm volatile("s_wait_asynccnt 0x0" ::: "memory");
#endif
}
// async loads complete in order: waiting <=3 releases once the previous
// tile's 3 copies have landed while the newest 3 remain in flight.
__device__ __forceinline__ void wait_async3() {
#if __has_builtin(__builtin_amdgcn_s_wait_asynccnt)
    __builtin_amdgcn_s_wait_asynccnt(3);
#else
    asm volatile("s_wait_asynccnt 0x3" ::: "memory");
#endif
}

// ---------------- fp32 -> bf16 weight conversion + transpose ----------------
// src: [K][N] f32 row-major  ->  dst: [N][K] bf16 row-major
__global__ void cvt_t_kernel(const float* __restrict__ src,
                             __bf16* __restrict__ dst, int Kd, int Nd) {
    int i = blockIdx.x * blockDim.x + threadIdx.x;
    if (i < Kd * Nd) {
        int k = i / Nd;
        int n = i - k * Nd;
        dst[(size_t)n * Kd + k] = (__bf16)src[i];
    }
}

// ---------------- LayerNorm: f32 [rows,1024] -> bf16 ----------------
__global__ __launch_bounds__(256) void ln_kernel(const float* __restrict__ in,
                                                 const float* __restrict__ g,
                                                 const float* __restrict__ bta,
                                                 __bf16* __restrict__ out) {
    int row = blockIdx.x;
    int tid = threadIdx.x;
    const float* rp = in + (size_t)row * TC;
    float4 v = ((const float4*)rp)[tid];
    float s  = v.x + v.y + v.z + v.w;
    float ss = v.x * v.x + v.y * v.y + v.z * v.z + v.w * v.w;
    #pragma unroll
    for (int off = 16; off > 0; off >>= 1) {
        s  += __shfl_xor(s, off, 32);
        ss += __shfl_xor(ss, off, 32);
    }
    __shared__ float sh_s[8], sh_ss[8];
    int wid = tid >> 5, lane = tid & 31;
    if (lane == 0) { sh_s[wid] = s; sh_ss[wid] = ss; }
    __syncthreads();
    if (tid == 0) {
        float ts = 0.f, tss = 0.f;
        #pragma unroll
        for (int i = 0; i < 8; ++i) { ts += sh_s[i]; tss += sh_ss[i]; }
        sh_s[0] = ts; sh_ss[0] = tss;
    }
    __syncthreads();
    float mean = sh_s[0] * (1.0f / TC);
    float var  = sh_ss[0] * (1.0f / TC) - mean * mean;
    float rinv = rsqrtf(var + 1e-5f);
    int base = tid * 4;
    float4 gg = ((const float4*)g)[tid];
    float4 bb = ((const float4*)bta)[tid];
    __bf16* op = out + (size_t)row * TC + base;
    op[0] = (__bf16)((v.x - mean) * rinv * gg.x + bb.x);
    op[1] = (__bf16)((v.y - mean) * rinv * gg.y + bb.y);
    op[2] = (__bf16)((v.z - mean) * rinv * gg.z + bb.z);
    op[3] = (__bf16)((v.w - mean) * rinv * gg.w + bb.w);
}

// ---------------- generic bf16 WMMA GEMM ----------------
// C[M,N] = A[M,K] (row major) * B[K,N], where B is provided TRANSPOSED as
// BT[N][K] row-major. f32 accumulate. Workgroup tile 64x128, 8 waves as 2x4,
// each wave 32x32 (2x2 WMMA tiles). Double-buffered async LDS staging.
#define GEMM_QKV  0
#define GEMM_RES  1
#define GEMM_GELU 2

__global__ __launch_bounds__(256) void gemm_bf16_kernel(
    const __bf16* __restrict__ A, const __bf16* __restrict__ BT,
    int M, int N, int K, int mode,
    float* __restrict__ outF, const float* __restrict__ res,
    __bf16* __restrict__ outQ, __bf16* __restrict__ outK,
    __bf16* __restrict__ outV, __bf16* __restrict__ outB) {
    __shared__ alignas(16) __bf16 As[2][64 * 32];    // m-major [64][32]
    __shared__ alignas(16) __bf16 Bs[2][128 * 32];   // n-major [128][32]

    int tid  = threadIdx.x;
    int lane = tid & 31;
    int wave = tid >> 5;
    int wm = wave >> 2;          // 0..1 (M)
    int wn = wave & 3;           // 0..3 (N)
    int mBase = blockIdx.y * 64;
    int nBase = blockIdx.x * 128;

    v8f zero = {};
    v8f acc[2][2];
    acc[0][0] = zero; acc[0][1] = zero; acc[1][0] = zero; acc[1][1] = zero;

    int la_idx = tid * 8;                 // A tile: 64x32 = 2048 elems, 16B/thread
    int la_r = la_idx >> 5, la_c = la_idx & 31;
    int bn = tid >> 1;                    // B tile: 128x32 = 4096 elems, 32B/thread
    int bk = (tid & 1) * 16;

    int kb  = ((lane >> 4) & 1) * 8;      // A frag K base (per half-wave)
    int kb2 = ((lane >> 4) & 1) * 16;     // unused (B frags use kb, same as A)
    int nl  = lane & 15;

    const __bf16* ap = &A[(size_t)(mBase + la_r) * K + la_c];
    const __bf16* bp = &BT[(size_t)(nBase + bn) * K + bk];

    int nTiles = K >> 5;
    // prologue: stage tile 0 into buffer 0
    async_copy_b128(&As[0][la_idx],        ap);
    async_copy_b128(&Bs[0][bn * 32 + bk],     bp);
    async_copy_b128(&Bs[0][bn * 32 + bk + 8], bp + 8);

    for (int it = 0; it < nTiles; ++it) {
        int cur = it & 1;
        if (it + 1 < nTiles) {
            // stage tile it+1 into the other buffer (safe: all waves finished
            // reading it at the previous iteration's trailing barrier)
            int k1 = (it + 1) << 5;
            async_copy_b128(&As[cur ^ 1][la_idx],        ap + k1);
            async_copy_b128(&Bs[cur ^ 1][bn * 32 + bk],     bp + k1);
            async_copy_b128(&Bs[cur ^ 1][bn * 32 + bk + 8], bp + k1 + 8);
            wait_async3();   // tile `it` landed; tile it+1 still in flight
        } else {
            wait_async0();
        }
        __syncthreads();

        const __bf16* Ac = As[cur];
        const __bf16* Bc = Bs[cur];
        v16bf afrag[2], bfrag[2];
        #pragma unroll
        for (int i = 0; i < 2; ++i) {
            int row = wm * 32 + i * 16 + nl;
            ((uint4*)&afrag[i])[0] = *(const uint4*)&Ac[row * 32 + kb];
            ((uint4*)&afrag[i])[1] = *(const uint4*)&Ac[row * 32 + 16 + kb];
        }
        #pragma unroll
        for (int j = 0; j < 2; ++j) {
            int col = wn * 32 + j * 16 + nl;
            ((uint4*)&bfrag[j])[0] = *(const uint4*)&Bc[col * 32 + kb];
            ((uint4*)&bfrag[j])[1] = *(const uint4*)&Bc[col * 32 + 16 + kb];
        }
        #pragma unroll
        for (int i = 0; i < 2; ++i)
            #pragma unroll
            for (int j = 0; j < 2; ++j)
                acc[i][j] = __builtin_amdgcn_wmma_f32_16x16x32_bf16(
                    false, afrag[i], false, bfrag[j], (short)0, acc[i][j],
                    false, false);
        __syncthreads();   // all reads done before next iteration's async writes
    }

    // epilogue: lane holds (m = r + 8*hi, n = lane&15) per 16x16 tile
    int hi = (lane >> 4) & 1;
    #pragma unroll
    for (int i = 0; i < 2; ++i) {
        #pragma unroll
        for (int j = 0; j < 2; ++j) {
            #pragma unroll
            for (int r = 0; r < 8; ++r) {
                int mg = mBase + wm * 32 + i * 16 + hi * 8 + r;
                int ng = nBase + wn * 32 + j * 16 + nl;
                float val = acc[i][j][r];
                if (mode == GEMM_RES) {
                    size_t idx = (size_t)mg * N + ng;
                    outF[idx] = val + res[idx];
                } else if (mode == GEMM_GELU) {
                    outB[(size_t)mg * N + ng] = (__bf16)gelu_f(val);
                } else { // GEMM_QKV: scatter to q/k/v [B,H,T,hd]
                    int part = ng >> 10;
                    int w    = ng & 1023;
                    int head = w >> 6;
                    int d    = w & 63;
                    int bb   = mg >> 11;
                    int t    = mg & 2047;
                    size_t off = (((size_t)(bb * TH + head)) * TT + t) * THD + d;
                    __bf16* dst = (part == 0) ? outQ : ((part == 1) ? outK : outV);
                    dst[off] = (__bf16)val;
                }
            }
        }
    }
}

// ---------------- flash attention: one wave per (b, h, 16-query block) ------
__global__ __launch_bounds__(32) void attn_kernel(
    const __bf16* __restrict__ Qm, const __bf16* __restrict__ Km,
    const __bf16* __restrict__ Vm, __bf16* __restrict__ Y) {
    __shared__ alignas(16) __bf16 vT[64 * 32];   // V^T for one 32-key chunk
    __shared__ alignas(16) __bf16 Ps[16 * 32];   // P tile (re-swizzle buffer)

    int blk = blockIdx.x;
    int qb = blk & 127;               // T/16
    int h  = (blk >> 7) & 15;
    int b  = blk >> 11;
    int qBase = qb * 16;
    size_t bh = (size_t)(b * TH + h) * TT * THD;
    const __bf16* qp = Qm + bh;
    const __bf16* kp = Km + bh;
    const __bf16* vp = Vm + bh;

    int lane = threadIdx.x;
    int hi  = (lane >> 4) & 1;
    int nl  = lane & 15;
    int kb  = hi * 8;
    int kb2 = hi * 16;

    // Q fragments (A layout), K=64 split into two 32-chunks
    v16bf qf[2];
    {
        int row = qBase + nl;
        #pragma unroll
        for (int s = 0; s < 2; ++s) {
            ((uint4*)&qf[s])[0] = *(const uint4*)&qp[(size_t)row * 64 + s * 32 + kb];
            ((uint4*)&qf[s])[1] = *(const uint4*)&qp[(size_t)row * 64 + s * 32 + 16 + kb];
        }
    }

    v8f zero = {};
    v8f o[4]; o[0] = zero; o[1] = zero; o[2] = zero; o[3] = zero;
    float mi[8], li[8];
    #pragma unroll
    for (int r = 0; r < 8; ++r) { mi[r] = -3.0e38f; li[r] = 0.f; }

    for (int jb = 0; jb < qBase + 16; jb += 32) {
        __syncthreads();
        // stage V^T: vT[d*32 + key]
        {
            int key = jb + lane;
            #pragma unroll
            for (int d = 0; d < 64; ++d)
                vT[d * 32 + lane] = vp[(size_t)key * 64 + d];
        }
        // S = (Q K^T) for 32 keys -> two 16x16 tiles
        v8f sacc[2]; sacc[0] = zero; sacc[1] = zero;
        #pragma unroll
        for (int j2 = 0; j2 < 2; ++j2) {
            int key = jb + j2 * 16 + nl;
            #pragma unroll
            for (int s = 0; s < 2; ++s) {
                v16bf kf;
                ((uint4*)&kf)[0] = *(const uint4*)&kp[(size_t)key * 64 + s * 32 + kb];
                ((uint4*)&kf)[1] = *(const uint4*)&kp[(size_t)key * 64 + s * 32 + 16 + kb];
                sacc[j2] = __builtin_amdgcn_wmma_f32_16x16x32_bf16(
                    false, qf[s], false, kf, (short)0, sacc[j2], false, false);
            }
        }
        // scale + causal mask + online softmax (row stats per half-wave)
        float pv[2][8], mch[8];
        #pragma unroll
        for (int r = 0; r < 8; ++r) {
            int qIdx = qBase + hi * 8 + r;
            float s0 = sacc[0][r] * 0.125f;
            float s1 = sacc[1][r] * 0.125f;
            if (jb + nl > qIdx)      s0 = -3.0e38f;
            if (jb + 16 + nl > qIdx) s1 = -3.0e38f;
            pv[0][r] = s0; pv[1][r] = s1;
            float mc = fmaxf(s0, s1);
            #pragma unroll
            for (int off = 1; off < 16; off <<= 1)
                mc = fmaxf(mc, __shfl_xor(mc, off, 32));
            mch[r] = mc;
        }
        #pragma unroll
        for (int r = 0; r < 8; ++r) {
            float mnew  = fmaxf(mi[r], mch[r]);
            float alpha = __expf(mi[r] - mnew);
            mi[r] = mnew;
            float p0 = __expf(pv[0][r] - mnew);
            float p1 = __expf(pv[1][r] - mnew);
            pv[0][r] = p0; pv[1][r] = p1;
            float ls = p0 + p1;
            #pragma unroll
            for (int off = 1; off < 16; off <<= 1)
                ls += __shfl_xor(ls, off, 32);
            li[r] = li[r] * alpha + ls;
            #pragma unroll
            for (int t2 = 0; t2 < 4; ++t2) o[t2][r] *= alpha;
        }
        // write P (C layout) -> LDS, re-read as A layout
        #pragma unroll
        for (int j2 = 0; j2 < 2; ++j2)
            #pragma unroll
            for (int r = 0; r < 8; ++r)
                Ps[(hi * 8 + r) * 32 + j2 * 16 + nl] = (__bf16)pv[j2][r];
        __syncthreads();
        v16bf pf;
        ((uint4*)&pf)[0] = *(const uint4*)&Ps[nl * 32 + kb];
        ((uint4*)&pf)[1] = *(const uint4*)&Ps[nl * 32 + 16 + kb];
        // O += P @ V   (4 d-tiles of 16)
        #pragma unroll
        for (int t2 = 0; t2 < 4; ++t2) {
            int d = t2 * 16 + nl;
            v16bf vf;
            ((uint4*)&vf)[0] = *(const uint4*)&vT[d * 32 + kb];
            ((uint4*)&vf)[1] = *(const uint4*)&vT[d * 32 + 16 + kb];
            o[t2] = __builtin_amdgcn_wmma_f32_16x16x32_bf16(
                false, pf, false, vf, (short)0, o[t2], false, false);
        }
    }
    // write Y[b,t,h*64+d] (bf16)
    #pragma unroll
    for (int t2 = 0; t2 < 4; ++t2) {
        #pragma unroll
        for (int r = 0; r < 8; ++r) {
            int t = qBase + hi * 8 + r;
            int d = t2 * 16 + nl;
            Y[((size_t)(b * TT + t)) * TC + h * THD + d] = (__bf16)(o[t2][r] / li[r]);
        }
    }
}

// ---------------------------------------------------------------------------
extern "C" void kernel_launch(void* const* d_in, const int* in_sizes, int n_in,
                              void* d_out, int out_size, void* d_ws, size_t ws_size,
                              hipStream_t stream) {
    const float* x        = (const float*)d_in[0];
    const float* w_attn   = (const float*)d_in[1];
    const float* w_aproj  = (const float*)d_in[2];
    const float* w_fc     = (const float*)d_in[3];
    const float* w_mproj  = (const float*)d_in[4];
    const float* ln1_g    = (const float*)d_in[5];
    const float* ln1_b    = (const float*)d_in[6];
    const float* ln2_g    = (const float*)d_in[7];
    const float* ln2_b    = (const float*)d_in[8];
    float* out = (float*)d_out;

    char* ws = (char*)d_ws;
    __bf16* wb_attn  = (__bf16*)(ws + OFF_WB_ATTN);
    __bf16* wb_aproj = (__bf16*)(ws + OFF_WB_APROJ);
    __bf16* wb_fc    = (__bf16*)(ws + OFF_WB_FC);
    __bf16* wb_mproj = (__bf16*)(ws + OFF_WB_MPROJ);
    __bf16* h1       = (__bf16*)(ws + OFF_H1);
    __bf16* qbuf     = (__bf16*)(ws + OFF_Q);
    __bf16* kbuf     = (__bf16*)(ws + OFF_K);
    __bf16* vbuf     = (__bf16*)(ws + OFF_V);
    __bf16* ybuf     = (__bf16*)(ws + OFF_Y);
    float*  xres     = (float*)(ws + OFF_XRES);
    __bf16* h2       = (__bf16*)(ws + OFF_H2);
    __bf16* h3       = (__bf16*)(ws + OFF_H3);

    // 1) convert weights fp32 -> bf16, transposed to [N][K]
    {
        int n;
        n = TC * 3 * TC; cvt_t_kernel<<<n / 256, 256, 0, stream>>>(w_attn,  wb_attn,  TC, 3 * TC);
        n = TC * TC;     cvt_t_kernel<<<n / 256, 256, 0, stream>>>(w_aproj, wb_aproj, TC, TC);
        n = TC * 4 * TC; cvt_t_kernel<<<n / 256, 256, 0, stream>>>(w_fc,    wb_fc,    TC, 4 * TC);
        n = 4 * TC * TC; cvt_t_kernel<<<n / 256, 256, 0, stream>>>(w_mproj, wb_mproj, 4 * TC, TC);
    }
    // 2) LN1
    ln_kernel<<<TM, 256, 0, stream>>>(x, ln1_g, ln1_b, h1);
    // 3) qkv = h1 @ w_attn, scattered to q/k/v
    gemm_bf16_kernel<<<dim3(3 * TC / 128, TM / 64), 256, 0, stream>>>(
        h1, wb_attn, TM, 3 * TC, TC, GEMM_QKV,
        nullptr, nullptr, qbuf, kbuf, vbuf, nullptr);
    // 4) attention -> ybuf
    attn_kernel<<<TB * TH * (TT / 16), 32, 0, stream>>>(qbuf, kbuf, vbuf, ybuf);
    // 5) xres = x + ybuf @ w_attn_proj
    gemm_bf16_kernel<<<dim3(TC / 128, TM / 64), 256, 0, stream>>>(
        ybuf, wb_aproj, TM, TC, TC, GEMM_RES,
        xres, x, nullptr, nullptr, nullptr, nullptr);
    // 6) LN2
    ln_kernel<<<TM, 256, 0, stream>>>(xres, ln2_g, ln2_b, h2);
    // 7) h3 = gelu(h2 @ w_fc)
    gemm_bf16_kernel<<<dim3(4 * TC / 128, TM / 64), 256, 0, stream>>>(
        h2, wb_fc, TM, 4 * TC, TC, GEMM_GELU,
        nullptr, nullptr, nullptr, nullptr, nullptr, h3);
    // 8) out = xres + h3 @ w_mlp_proj
    gemm_bf16_kernel<<<dim3(TC / 128, TM / 64), 256, 0, stream>>>(
        h3, wb_mproj, TM, TC, 4 * TC, GEMM_RES,
        out, xres, nullptr, nullptr, nullptr, nullptr);
}
